// MambaBackbone_61804579389937
// MI455X (gfx1250) — matrix operand order repaired and checked by reading, not compiled, not run on hardware
//
#include <hip/hip_runtime.h>

// ---------------- problem constants ----------------
#define D_MODEL   1024
#define NUM_LAYERS 4
#define D_STATE   64
#define D_CONV    4
#define HEADDIM   64
#define D_INNER   2048
#define NHEADS    32
#define CONV_DIM  2176          // D_INNER + 2*D_STATE
#define D_IN_PROJ 4256          // 2*D_INNER + 2*D_STATE + NHEADS
#define NPAD_IN   4352          // D_IN_PROJ rounded up to 128 (padded Bt rows)
#define BATCH     4
#define SEQ       2048
#define NTOK      (BATCH*SEQ)   // 8192
#define EPS_F     1e-5f

#define USE_ASYNC 1             // gfx1250 GLOBAL_LOAD_ASYNC_TO_LDS path (set 0 to fall back)

typedef unsigned short u16;
typedef __attribute__((ext_vector_type(16))) __bf16 v16bf;
typedef __attribute__((ext_vector_type(8)))  float  v8f;

__device__ __forceinline__ u16 f2bf(float f) {
  unsigned u = __float_as_uint(f);
  return (u16)((u + 0x7FFFu + ((u >> 16) & 1u)) >> 16);
}
__device__ __forceinline__ float siluf_(float x) { return x / (1.0f + __expf(-x)); }

#if USE_ASYNC
// CDNA5 async global->LDS copies (tracked by ASYNCcnt). Generic LDS pointer's
// low 32 bits are the wave-relative LDS byte address (aperture in high bits).
__device__ __forceinline__ void async_ld_b128(void* lds, const void* gaddr) {
  unsigned l = (unsigned)(unsigned long long)lds;
  unsigned long long g = (unsigned long long)gaddr;
  asm volatile("global_load_async_to_lds_b128 %0, %1, off" :: "v"(l), "v"(g) : "memory");
}
__device__ __forceinline__ void async_ld_b64(void* lds, const void* gaddr) {
  unsigned l = (unsigned)(unsigned long long)lds;
  unsigned long long g = (unsigned long long)gaddr;
  asm volatile("global_load_async_to_lds_b64 %0, %1, off" :: "v"(l), "v"(g) : "memory");
}
__device__ __forceinline__ void wait_async0() {
  asm volatile("s_wait_asynccnt 0x0" ::: "memory");
}
#endif

// ---------------- weight cast + transpose: src[nmat][rows][cols] -> dst[nmat][npad][rows] (bf16)
__global__ void k_cast_transpose(const float* __restrict__ src, u16* __restrict__ dst,
                                 int rows, int cols, int npad, long long total) {
  long long idx = (long long)blockIdx.x * 256 + threadIdx.x;
  if (idx >= total) return;
  int c = (int)(idx % cols);
  long long t = idx / cols;
  int r = (int)(t % rows);
  int m = (int)(t / rows);
  dst[((long long)m * npad + c) * rows + r] = f2bf(src[idx]);
}

__global__ void k_copy_f32(const float* __restrict__ src, float* __restrict__ dst, long long n) {
  long long idx = (long long)blockIdx.x * 256 + threadIdx.x;
  if (idx < n) dst[idx] = src[idx];
}

// ---------------- LayerNorm over D_MODEL, output bf16 ----------------
__global__ void k_layernorm(const float* __restrict__ x, const float* __restrict__ w,
                            const float* __restrict__ b, u16* __restrict__ out) {
  __shared__ float red[256];
  const int tok = blockIdx.x;
  const float* xr = x + (long long)tok * D_MODEL;
  float v[4], s = 0.f, ss = 0.f;
#pragma unroll
  for (int i = 0; i < 4; i++) {
    v[i] = xr[threadIdx.x + i * 256];
    s += v[i]; ss += v[i] * v[i];
  }
  red[threadIdx.x] = s; __syncthreads();
  for (int o = 128; o > 0; o >>= 1) { if (threadIdx.x < o) red[threadIdx.x] += red[threadIdx.x + o]; __syncthreads(); }
  const float mu = red[0] * (1.0f / D_MODEL);
  __syncthreads();
  red[threadIdx.x] = ss; __syncthreads();
  for (int o = 128; o > 0; o >>= 1) { if (threadIdx.x < o) red[threadIdx.x] += red[threadIdx.x + o]; __syncthreads(); }
  const float var = red[0] * (1.0f / D_MODEL) - mu * mu;
  const float rstd = rsqrtf(var + EPS_F);
#pragma unroll
  for (int i = 0; i < 4; i++) {
    int c = threadIdx.x + i * 256;
    out[(long long)tok * D_MODEL + c] = f2bf((v[i] - mu) * rstd * w[c] + b[c]);
  }
}

// ---------------- bf16 WMMA GEMM: C[M,N] (f32) = A[M,K](bf16,rowmaj) * Bt[Npad,K](bf16)^T --------
// Block tile 128x128, 8 waves (4x2), wave tile 32x64 via 2x4 v_wmma_f32_16x16x32_bf16.
// Bt must have >= ceil(N/128)*128 valid (readable) rows; stores are guarded by N.
// If R != nullptr: C = R + A*Bt^T (fused residual; block owns its tile -> race-free).
__launch_bounds__(256)
__global__ void k_gemm_bf16(const u16* __restrict__ A, const u16* __restrict__ Bt,
                            float* __restrict__ C, const float* __restrict__ R,
                            int M, int N, int K) {
  __shared__ u16 lds_a[128 * 32];
  __shared__ u16 lds_b[128 * 32];
  const int tid  = threadIdx.x;
  const int lane = tid & 31;
  const int wave = tid >> 5;
  const int wm = wave & 3, wn = wave >> 2;        // 4x2 wave grid: 32 rows x 64 cols each
  const int hl = lane >> 4, mrow = lane & 15;     // half-wave, row-in-16
  const int blockN = blockIdx.x * 128;
  const int blockM = blockIdx.y * 128;

  v8f acc[2][4];
#pragma unroll
  for (int i = 0; i < 2; i++)
#pragma unroll
    for (int j = 0; j < 4; j++) {
      v8f z = {0.f, 0.f, 0.f, 0.f, 0.f, 0.f, 0.f, 0.f};
      acc[i][j] = z;
    }

  // staging: both tiles 128 rows x 32 cols bf16; thread -> (row, 16-element half-row)
  const int sr = tid >> 1, sh = tid & 1;
  const u16* aPtr = A  + (long long)(blockM + sr) * K + sh * 16;
  const u16* bPtr = Bt + (long long)(blockN + sr) * K + sh * 16;
  u16* lA = lds_a + sr * 32 + sh * 16;
  u16* lB = lds_b + sr * 32 + sh * 16;

  for (int k0 = 0; k0 < K; k0 += 32) {
    __syncthreads();                      // previous compute finished reading LDS
#if USE_ASYNC
    async_ld_b128(lA,     aPtr + k0);
    async_ld_b128(lA + 8, aPtr + k0 + 8);
    async_ld_b128(lB,     bPtr + k0);
    async_ld_b128(lB + 8, bPtr + k0 + 8);
#else
    {
      const uint4* ga = (const uint4*)(aPtr + k0);
      ((uint4*)lA)[0] = ga[0]; *(uint4*)(lA + 8) = ga[1];
      const uint4* gb = (const uint4*)(bPtr + k0);
      ((uint4*)lB)[0] = gb[0]; *(uint4*)(lB + 8) = gb[1];
    }
#endif
    if (k0 + 32 < K) {                    // gfx1250 global_prefetch_b8 for next K slice
      __builtin_prefetch(aPtr + k0 + 32, 0, 1);
      __builtin_prefetch(bPtr + k0 + 32, 0, 1);
    }
#if USE_ASYNC
    wait_async0();
#endif
    __syncthreads();                      // tile resident for all waves

    union Frag { v16bf v; uint4 q[2]; };
    Frag a[2], b[4];
#pragma unroll
    for (int mi = 0; mi < 2; mi++) {      // A lane<16: K{0..7,16..23}; lane>=16: K{8..15,24..31}
      int base = (wm * 32 + mi * 16 + mrow) * 32 + (hl ? 8 : 0);
      a[mi].q[0] = *(const uint4*)(lds_a + base);
      a[mi].q[1] = *(const uint4*)(lds_a + base + 16);
    }
#pragma unroll
    for (int ni = 0; ni < 4; ni++) {      // B lane<16: K{0..7,16..23}; lane>=16: K{8..15,24..31}
      int base = (wn * 64 + ni * 16 + mrow) * 32 + (hl ? 8 : 0);
      b[ni].q[0] = *(const uint4*)(lds_b + base);
      b[ni].q[1] = *(const uint4*)(lds_b + base + 16);
    }
#pragma unroll
    for (int mi = 0; mi < 2; mi++)
#pragma unroll
      for (int ni = 0; ni < 4; ni++)
        acc[mi][ni] = __builtin_amdgcn_wmma_f32_16x16x32_bf16(
            false, a[mi].v, false, b[ni].v, (short)0, acc[mi][ni], false, false);
  }

#pragma unroll
  for (int mi = 0; mi < 2; mi++)
#pragma unroll
    for (int ni = 0; ni < 4; ni++) {
#pragma unroll
      for (int v = 0; v < 8; v++) {       // D VGPR v: M = v + 8*hl, N = mrow
        int gm = blockM + wm * 32 + mi * 16 + hl * 8 + v;
        int gn = blockN + wn * 64 + ni * 16 + mrow;
        if (gm < M && gn < N) {
          long long idx = (long long)gm * N + gn;
          float val = acc[mi][ni][v];
          C[idx] = R ? (R[idx] + val) : val;
        }
      }
    }
}

// ---------------- causal depthwise conv(4) + SiLU over xBC slice of zxbcdt ----------------
__global__ void k_conv(const float* __restrict__ zx, const float* __restrict__ cw,
                       const float* __restrict__ cb, float* __restrict__ out) {
  long long idx = (long long)blockIdx.x * 256 + threadIdx.x;
  if (idx >= (long long)NTOK * CONV_DIM) return;
  int c  = (int)(idx % CONV_DIM);
  long long bl = idx / CONV_DIM;
  int l = (int)(bl & (SEQ - 1));
  float acc = cb[c];
#pragma unroll
  for (int k = 0; k < D_CONV; k++) {
    int dl = l - (D_CONV - 1) + k;
    if (dl >= 0)
      acc += cw[k * CONV_DIM + c] *
             zx[(bl - (D_CONV - 1) + k) * D_IN_PROJ + D_INNER + c];
  }
  out[idx] = siluf_(acc);
}

// ---------------- dt = softplus(zx[..., -NHEADS:] + dt_bias) ----------------
__global__ void k_dt(const float* __restrict__ zx, const float* __restrict__ dt_bias,
                     float* __restrict__ dt) {
  int idx = blockIdx.x * 256 + threadIdx.x;
  if (idx >= NTOK * NHEADS) return;
  int h  = idx & (NHEADS - 1);
  int bl = idx >> 5;
  float v = zx[(long long)bl * D_IN_PROJ + (D_IN_PROJ - NHEADS) + h] + dt_bias[h];
  dt[idx] = (v > 20.f) ? v : log1pf(__expf(v));
}

// ---------------- selective scan: 1 block per (b,head), thread p owns h[p][0..63] in regs.
// Groups of 8 timesteps; B/C staged straight into double-buffered LDS with async copies;
// one barrier per group; xs/dt pipelined through registers.
#define SCAN_T 8
#define SCAN_NG (SEQ / SCAN_T)   // 256
__launch_bounds__(64)
__global__ void k_scan(const float* __restrict__ xbc, const float* __restrict__ dt,
                       const float* __restrict__ A_log, const float* __restrict__ Dp,
                       float* __restrict__ y) {
  __shared__ float bc[2][SCAN_T][2 * D_STATE];   // B (0..63) | C (64..127) per step
  const int h = blockIdx.x & (NHEADS - 1);
  const int b = blockIdx.x >> 5;
  const int p = threadIdx.x;                     // 0..63
  const float A  = -__expf(A_log[h]);
  const float Dv = Dp[h];
  float hreg[D_STATE];
#pragma unroll
  for (int n = 0; n < D_STATE; n++) hreg[n] = 0.f;

  const float* xbase = xbc + (long long)b * SEQ * CONV_DIM;
  const float* dbase = dt + (long long)b * SEQ * NHEADS;
  float* ybase = y + (long long)b * SEQ * D_INNER;

  float rXS[SCAN_T]; float rDT[SCAN_T];
#if !USE_ASYNC
  float2 rBC[SCAN_T];
#endif

  // prologue: group 0
  {
#pragma unroll
    for (int tt = 0; tt < SCAN_T; tt++) {
      const float* xb = xbase + (long long)tt * CONV_DIM;
#if USE_ASYNC
      async_ld_b64(&bc[0][tt][2 * p], xb + D_INNER + 2 * p);
#else
      rBC[tt] = *(const float2*)(xb + D_INNER + 2 * p);
#endif
      rXS[tt] = xb[h * HEADDIM + p];
      rDT[tt] = dbase[(long long)tt * NHEADS + h];
    }
  }

  for (int g = 0; g < SCAN_NG; g++) {
    const int buf = g & 1;
#if !USE_ASYNC
#pragma unroll
    for (int tt = 0; tt < SCAN_T; tt++)
      *(float2*)&bc[buf][tt][2 * p] = rBC[tt];
#endif
    float xs_c[SCAN_T], dt_c[SCAN_T];
#pragma unroll
    for (int tt = 0; tt < SCAN_T; tt++) { xs_c[tt] = rXS[tt]; dt_c[tt] = rDT[tt]; }
#if USE_ASYNC
    wait_async0();                       // this wave's group-g B/C resident in LDS
#endif
    __syncthreads();                     // whole block sees bc[buf]

    // issue next group's transfers (consumed next iteration)
    if (g + 1 < SCAN_NG) {
      const long long tok0 = (long long)(g + 1) * SCAN_T;
#pragma unroll
      for (int tt = 0; tt < SCAN_T; tt++) {
        const float* xb = xbase + (tok0 + tt) * CONV_DIM;
#if USE_ASYNC
        async_ld_b64(&bc[buf ^ 1][tt][2 * p], xb + D_INNER + 2 * p);
#else
        rBC[tt] = *(const float2*)(xb + D_INNER + 2 * p);
#endif
        rXS[tt] = xb[h * HEADDIM + p];
        rDT[tt] = dbase[(tok0 + tt) * NHEADS + h];
      }
    }

    // compute the 8 steps of this group
#pragma unroll
    for (int tt = 0; tt < SCAN_T; tt++) {
      const float dtv = dt_c[tt];
      const float dA  = __expf(dtv * A);
      const float xs  = xs_c[tt];
      const float dBx = dtv * xs;
      const float* Brow = &bc[buf][tt][0];
      const float* Crow = &bc[buf][tt][D_STATE];
      float y0 = 0.f, y1 = 0.f;
#pragma unroll
      for (int n = 0; n < D_STATE; n += 2) {
        hreg[n]     = hreg[n]     * dA + dBx * Brow[n];
        y0 += hreg[n]     * Crow[n];
        hreg[n + 1] = hreg[n + 1] * dA + dBx * Brow[n + 1];
        y1 += hreg[n + 1] * Crow[n + 1];
      }
      ybase[(long long)(g * SCAN_T + tt) * D_INNER + h * HEADDIM + p] = y0 + y1 + Dv * xs;
    }
  }
}

// ---------------- y *= silu(z); RMSNorm(D_INNER)*norm_w; -> bf16 ----------------
__global__ void k_gate_norm(const float* __restrict__ y, const float* __restrict__ zx,
                            const float* __restrict__ nw, u16* __restrict__ out) {
  __shared__ float red[256];
  const int tok = blockIdx.x;
  const float* yr = y + (long long)tok * D_INNER;
  const float* zr = zx + (long long)tok * D_IN_PROJ;   // z = zx[..., :D_INNER]
  float v[8]; float ss = 0.f;
#pragma unroll
  for (int i = 0; i < 8; i++) {
    int c = threadIdx.x + i * 256;
    float val = yr[c] * siluf_(zr[c]);
    v[i] = val; ss += val * val;
  }
  red[threadIdx.x] = ss; __syncthreads();
  for (int o = 128; o > 0; o >>= 1) { if (threadIdx.x < o) red[threadIdx.x] += red[threadIdx.x + o]; __syncthreads(); }
  const float scale = rsqrtf(red[0] * (1.0f / D_INNER) + EPS_F);
#pragma unroll
  for (int i = 0; i < 8; i++) {
    int c = threadIdx.x + i * 256;
    out[(long long)tok * D_INNER + c] = f2bf(v[i] * scale * nw[c]);
  }
}

// ---------------- host launcher ----------------
extern "C" void kernel_launch(void* const* d_in, const int* in_sizes, int n_in,
                              void* d_out, int out_size, void* d_ws, size_t ws_size,
                              hipStream_t stream) {
  (void)in_sizes; (void)n_in; (void)out_size; (void)ws_size;
  const float* x_in    = (const float*)d_in[0];
  const float* ln_w    = (const float*)d_in[1];
  const float* ln_b    = (const float*)d_in[2];
  const float* in_w    = (const float*)d_in[3];
  const float* conv_w  = (const float*)d_in[4];
  const float* conv_b  = (const float*)d_in[5];
  const float* dt_bias = (const float*)d_in[6];
  const float* A_log   = (const float*)d_in[7];
  const float* Dp      = (const float*)d_in[8];
  const float* norm_w  = (const float*)d_in[9];
  const float* out_w   = (const float*)d_in[10];

  size_t off = 0;
  auto take = [&](size_t bytes) -> void* {
    off = (off + 255) & ~(size_t)255;
    void* p = (char*)d_ws + off;
    off += bytes;
    return p;
  };
  const size_t wt_in_bytes = (size_t)NUM_LAYERS * NPAD_IN * D_MODEL * 2;
  u16*   wt_in  = (u16*)  take(wt_in_bytes);                                  // [l][NPAD_IN][d]
  u16*   wt_out = (u16*)  take((size_t)NUM_LAYERS * D_MODEL * D_INNER * 2);   // [l][d][e]
  float* x_cur  = (float*)take((size_t)NTOK * D_MODEL * 4);
  u16*   xn     = (u16*)  take((size_t)NTOK * D_MODEL * 2);
  float* zx     = (float*)take((size_t)NTOK * D_IN_PROJ * 4);
  float* xbc    = (float*)take((size_t)NTOK * CONV_DIM * 4);
  float* dtb    = (float*)take((size_t)NTOK * NHEADS * 4);
  float* yb     = (float*)take((size_t)NTOK * D_INNER * 4);
  u16*   ybf    = (u16*)  take((size_t)NTOK * D_INNER * 2);

  // one-time (per launch) weight prep: cast to bf16 + transpose to [Npad][K]
  {
    hipMemsetAsync(wt_in, 0, wt_in_bytes, stream);   // zero padded rows (capture-safe)
    long long t1 = (long long)NUM_LAYERS * D_MODEL * D_IN_PROJ;
    k_cast_transpose<<<(unsigned)((t1 + 255) / 256), 256, 0, stream>>>(
        in_w, wt_in, D_MODEL, D_IN_PROJ, NPAD_IN, t1);
    long long t2 = (long long)NUM_LAYERS * D_INNER * D_MODEL;
    k_cast_transpose<<<(unsigned)((t2 + 255) / 256), 256, 0, stream>>>(
        out_w, wt_out, D_INNER, D_MODEL, D_MODEL, t2);
    long long nx = (long long)NTOK * D_MODEL;
    k_copy_f32<<<(unsigned)((nx + 255) / 256), 256, 0, stream>>>(x_in, x_cur, nx);
  }

  for (int i = 0; i < NUM_LAYERS; i++) {
    k_layernorm<<<NTOK, 256, 0, stream>>>(x_cur, ln_w + (size_t)i * D_MODEL,
                                          ln_b + (size_t)i * D_MODEL, xn);

    // zxbcdt = xn @ in_w[i]   (M=8192, N=4256 (padded rows readable), K=1024)
    k_gemm_bf16<<<dim3(NPAD_IN / 128, NTOK / 128), 256, 0, stream>>>(
        xn, wt_in + (size_t)i * NPAD_IN * D_MODEL, zx, nullptr,
        NTOK, D_IN_PROJ, D_MODEL);

    k_conv<<<(unsigned)(((long long)NTOK * CONV_DIM + 255) / 256), 256, 0, stream>>>(
        zx, conv_w + (size_t)i * D_CONV * CONV_DIM, conv_b + (size_t)i * CONV_DIM, xbc);

    k_dt<<<(NTOK * NHEADS) / 256, 256, 0, stream>>>(zx, dt_bias + (size_t)i * NHEADS, dtb);

    k_scan<<<BATCH * NHEADS, 64, 0, stream>>>(xbc, dtb, A_log + (size_t)i * NHEADS,
                                              Dp + (size_t)i * NHEADS, yb);

    k_gate_norm<<<NTOK, 256, 0, stream>>>(yb, zx, norm_w + (size_t)i * D_INNER, ybf);

    // x = x + y @ out_w[i]   (M=8192, N=1024, K=2048); last layer writes d_out directly
    float* Cdst = (i == NUM_LAYERS - 1) ? (float*)d_out : x_cur;
    k_gemm_bf16<<<dim3(D_MODEL / 128, NTOK / 128), 256, 0, stream>>>(
        ybf, wt_out + (size_t)i * D_MODEL * D_INNER, Cdst, x_cur,
        NTOK, D_MODEL, D_INNER);
  }
}
